// FIVEMPNN_80238579024175
// MI455X (gfx1250) — matrix operand order repaired and checked by reading, not compiled, not run on hardware
//
#include <hip/hip_runtime.h>
#include <hip/hip_bf16.h>

// ---------------------------------------------------------------------------
// 5-layer MPNN for MI455X (gfx1250, wave32, WMMA + TDM).
//   per layer:  h = bf16(x) @ Wl + bl            (WMMA bf16 GEMM, fp32 accum)
//               m[dst] += h[src]                 (fp32 atomics, wave/edge)
//               x = leaky( [h|m] @ Wu + bu )     (WMMA bf16 GEMM, K=256)
//   then mean-pool per graph + 128->2 head + log_softmax.
// Weights are staged into LDS by the Tensor Data Mover (tensor_load_to_lds)
// with hardware row padding (K+8 bf16 stride) to avoid LDS bank conflicts.
// ---------------------------------------------------------------------------

typedef __attribute__((ext_vector_type(16))) __bf16       v16bf;
typedef __attribute__((ext_vector_type(8)))  __bf16       v8bf;
typedef __attribute__((ext_vector_type(8)))  float        v8f;
typedef __attribute__((ext_vector_type(4)))  unsigned int v4u;
typedef __attribute__((ext_vector_type(8)))  unsigned int v8u;

#define HIDC 128

// ---------------- small utility kernels ----------------

__global__ void zero_f32_kernel(float* __restrict__ p, int n) {
    int i = blockIdx.x * 256 + threadIdx.x;
    if (i < n) p[i] = 0.0f;
}

__global__ void f32_to_bf16_kernel(const float* __restrict__ s,
                                   __bf16* __restrict__ d, int n) {
    int i = blockIdx.x * 256 + threadIdx.x;
    if (i < n) d[i] = (__bf16)s[i];
}

// dst[n*K + k] = (bf16) src[k*N + n]   (weight transpose to [N][K] bf16)
__global__ void transpose_bf16_kernel(const float* __restrict__ src,
                                      __bf16* __restrict__ dst, int K, int N) {
    int i = blockIdx.x * 256 + threadIdx.x;
    if (i >= K * N) return;
    int k = i / N, n = i % N;
    dst[n * K + k] = (__bf16)src[k * N + n];
}

// ---------------- WMMA GEMM:  out = leaky?( A @ W + bias ) ----------------
// A is [M][128] bf16; for K==256 the second 128 K-columns come from A1
// (the "concat" operand). WT is the transposed weight [N=128][K] bf16.
// TDM DMAs WT into LDS with a hardware-padded row stride of K+8 bf16
// (pad_amount = 4 dwords after every K/2-dword row) to kill bank conflicts.

__global__ __launch_bounds__(256)
void mpnn_gemm_wmma_kernel(const __bf16* __restrict__ A0,
                           const __bf16* __restrict__ A1,
                           const __bf16* __restrict__ WT,
                           const float*  __restrict__ bias,
                           float*  __restrict__ outF,
                           __bf16* __restrict__ outB,
                           int M, int K, int leaky)
{
    const int KP = K + 8;                    // padded LDS row stride (bf16)
    extern __shared__ __bf16 lds[];          // [128][KP]
    const int tid  = threadIdx.x;
    const int wave = tid >> 5;
    const int lane = tid & 31;

    // ---- Tensor Data Mover: DMA transposed weights global -> LDS ----
    if (wave == 0) {
        const unsigned long long ga = (unsigned long long)(size_t)WT;
        const unsigned ldsAddr = (unsigned)(size_t)(&lds[0]);   // LDS byte offset
        const unsigned rowDw = (unsigned)(K >> 1);              // row len in dwords
        const unsigned padIvl = (K == 128) ? 5u : 6u;           // 64 / 128 dwords
        // D# group 0: count=1 (valid), lds_addr, 57-bit global addr, type=2
        v4u g0;
        g0[0] = 1u;
        g0[1] = ldsAddr;
        g0[2] = (unsigned)(ga & 0xffffffffu);
        g0[3] = (unsigned)((ga >> 32) & 0x01ffffffu) | (2u << 30);
        // D# group 1: data_size=4B(code 2), pad_enable, pad_interval, pad_amount=4dw,
        //             tensor_dim0 = tile_dim0 = K/2 dwords, tensor_dim1 = tile_dim1 = 128,
        //             tensor_dim0_stride = K/2 (contiguous rows)
        v8u g1;
        g1[0] = (2u << 16) | (1u << 20) | (padIvl << 22) | (3u << 25);
        g1[1] = rowDw << 16;        // tensor_dim0[15:0] in bits[63:48]
        g1[2] = 128u  << 16;        // tensor_dim1 = 128  in bits[111:80]
        g1[3] = rowDw << 16;        // tile_dim0          in bits[127:112]
        g1[4] = 128u;               // tile_dim1 = 128    in bits[143:128]
        g1[5] = rowDw;              // tensor_dim0_stride in bits[207:160]
        g1[6] = 0u;
        g1[7] = 0u;
        asm volatile("tensor_load_to_lds %0, %1"
                     :: "s"(g0), "s"(g1)
                     : "memory");
        __builtin_amdgcn_s_wait_tensorcnt(0);
    }
    __syncthreads();

    const int rowTile = blockIdx.x * 8 + wave;   // one 16-row strip per wave
    if (rowTile * 16 >= M) return;               // wave-uniform exit
    const int rowBase = rowTile * 16;
    const int rA  = lane & 15;                   // M index (A) / N index (B,C)
    const int hiA = lane >> 4;                   // lane-half select

    // accumulators: 8 N-tiles of 16x16, initialized with the bias row
    v8f acc[8];
    #pragma unroll
    for (int nt = 0; nt < 8; ++nt) {
        float bv = bias[nt * 16 + rA];
        #pragma unroll
        for (int r = 0; r < 8; ++r) acc[nt][r] = bv;
    }

    const __bf16* aRow0 = A0 + (size_t)(rowBase + rA) * HIDC;
    const __bf16* aRow1 = A1 ? (A1 + (size_t)(rowBase + rA) * HIDC) : aRow0;

    const int nk = K >> 5;                       // K-chunks of 32
    for (int kc = 0; kc < nk; ++kc) {
        const __bf16* ap = (kc < 4) ? aRow0 : aRow1;   // concat operand
        const int kb = ((kc & 3) << 5) + (hiA << 3);
        // A fragment, per ISA 16-bit A 16x32 layout:
        //   elems 0..7  = A[row][kb+0..7], elems 8..15 = A[row][kb+16..23]
        v8bf alo = *(const v8bf*)(ap + kb);
        v8bf ahi = *(const v8bf*)(ap + kb + 16);
        if (kc + 1 < nk) __builtin_prefetch((const void*)(ap + kb + 32), 0, 0);
        v16bf af;
        #pragma unroll
        for (int i = 0; i < 8; ++i) { af[i] = alo[i]; af[8 + i] = ahi[i]; }

        const int kbB = (kc << 5) + (hiA << 3);  // B: K = kbB + 0..7, kbB + 16..23
        #pragma unroll
        for (int nt = 0; nt < 8; ++nt) {
            const int col = nt * 16 + rA;
            const __bf16* bp = lds + col * KP + kbB;
            v8bf blo = *(const v8bf*)(bp);
            v8bf bhi = *(const v8bf*)(bp + 16);
            v16bf bf2;
            #pragma unroll
            for (int i = 0; i < 8; ++i) { bf2[i] = blo[i]; bf2[8 + i] = bhi[i]; }
            acc[nt] = __builtin_amdgcn_wmma_f32_16x16x32_bf16(
                false, af, false, bf2, (short)0, acc[nt], false, false);
        }
    }

    // store: VGPR r -> row rowBase + r + 8*hiA, col = nt*16 + rA
    #pragma unroll
    for (int r = 0; r < 8; ++r) {
        const int row = rowBase + r + hiA * 8;
        #pragma unroll
        for (int nt = 0; nt < 8; ++nt) {
            float z = acc[nt][r];
            if (leaky) z = (z >= 0.0f) ? z : 0.1f * z;
            size_t off = (size_t)row * HIDC + nt * 16 + rA;
            outF[off] = z;
            outB[off] = (__bf16)z;
        }
    }
}

// ---------------- edge scatter-add: m[dst] += h[src] ----------------
// one wave per edge, float4 per lane -> coalesced 512B row moves + f32 atomics

__global__ __launch_bounds__(256)
void edge_scatter_kernel(const float* __restrict__ h,
                         const int* __restrict__ srcIdx,
                         const int* __restrict__ dstIdx,
                         float* __restrict__ m, int nE)
{
    int w    = (blockIdx.x * 256 + threadIdx.x) >> 5;
    int lane = threadIdx.x & 31;
    if (w >= nE) return;
    int s = srcIdx[w];
    int d = dstIdx[w];
    const float4 v = ((const float4*)(h + (size_t)s * HIDC))[lane];
    float* mp = m + (size_t)d * HIDC + lane * 4;
    atomicAdd(mp + 0, v.x);
    atomicAdd(mp + 1, v.y);
    atomicAdd(mp + 2, v.z);
    atomicAdd(mp + 3, v.w);
}

// ---------------- mean-pool (segment sum + counts) ----------------

__global__ __launch_bounds__(256)
void pool_kernel(const float* __restrict__ x, const int* __restrict__ batch,
                 float* __restrict__ pooled, float* __restrict__ cnt, int nNodes)
{
    int w    = (blockIdx.x * 256 + threadIdx.x) >> 5;
    int lane = threadIdx.x & 31;
    if (w >= nNodes) return;
    int g = batch[w];
    const float4 v = ((const float4*)(x + (size_t)w * HIDC))[lane];
    float* pp = pooled + (size_t)g * HIDC + lane * 4;
    atomicAdd(pp + 0, v.x);
    atomicAdd(pp + 1, v.y);
    atomicAdd(pp + 2, v.z);
    atomicAdd(pp + 3, v.w);
    if (lane == 0) atomicAdd(cnt + g, 1.0f);
}

// ---------------- head: mean, 128->2 linear, log_softmax ----------------

__global__ void head_kernel(const float* __restrict__ pooled,
                            const float* __restrict__ cnt,
                            const float* __restrict__ Wout,
                            const float* __restrict__ bout,
                            float* __restrict__ out, int nGraphs)
{
    int g = blockIdx.x * 256 + threadIdx.x;
    if (g >= nGraphs) return;
    float inv = 1.0f / fmaxf(cnt[g], 1.0f);
    float a0 = bout[0], a1 = bout[1];
    for (int k = 0; k < HIDC; ++k) {
        float p = pooled[(size_t)g * HIDC + k] * inv;
        a0 += p * Wout[k * 2 + 0];
        a1 += p * Wout[k * 2 + 1];
    }
    float mx = fmaxf(a0, a1);
    float e0 = __expf(a0 - mx), e1 = __expf(a1 - mx);
    float lse = __logf(e0 + e1);
    out[g * 2 + 0] = (a0 - mx) - lse;
    out[g * 2 + 1] = (a1 - mx) - lse;
}

// ---------------------------------------------------------------------------

extern "C" void kernel_launch(void* const* d_in, const int* in_sizes, int n_in,
                              void* d_out, int out_size, void* d_ws, size_t ws_size,
                              hipStream_t stream)
{
    (void)n_in; (void)out_size; (void)ws_size;

    const float* x     = (const float*)d_in[0];
    const int*   edges = (const int*)d_in[1];    // [2][nE]
    const int*   batch = (const int*)d_in[2];

    const int nNodes  = in_sizes[0] / HIDC;      // 100000
    const int nEdges  = in_sizes[1] / 2;         // 1600000
    const int nGraphs = 256;

    const float *Wl[5], *bl[5], *Wu[5], *bu[5];
    for (int i = 0; i < 5; ++i) {
        Wl[i] = (const float*)d_in[3 + 4 * i];   // [128][128]
        bl[i] = (const float*)d_in[4 + 4 * i];   // [128]
        Wu[i] = (const float*)d_in[5 + 4 * i];   // [256][128]
        bu[i] = (const float*)d_in[6 + 4 * i];   // [128]
    }
    const float* Wout = (const float*)d_in[23];  // [128][2]
    const float* bout = (const float*)d_in[24];  // [2]

    // ---- workspace carve-out (256B aligned) ----
    char* ws = (char*)d_ws;
    size_t off = 0;
    #define WS_ALLOC(ty, nelem) \
        (ty*)(ws + off); off += (((size_t)(nelem) * sizeof(ty)) + 255) & ~(size_t)255
    __bf16* xb = WS_ALLOC(__bf16, (size_t)nNodes * HIDC);
    float*  hF = WS_ALLOC(float,  (size_t)nNodes * HIDC);
    __bf16* hB = WS_ALLOC(__bf16, (size_t)nNodes * HIDC);
    float*  mF = WS_ALLOC(float,  (size_t)nNodes * HIDC);
    __bf16* mB = WS_ALLOC(__bf16, (size_t)nNodes * HIDC);
    __bf16* wlT[5];
    __bf16* wuT[5];
    for (int i = 0; i < 5; ++i) {
        wlT[i] = WS_ALLOC(__bf16, 128 * 128);
        wuT[i] = WS_ALLOC(__bf16, 128 * 256);
    }
    float* pooled = WS_ALLOC(float, (size_t)nGraphs * HIDC);
    float* cnt    = WS_ALLOC(float, nGraphs);
    #undef WS_ALLOC

    const int nFeat   = nNodes * HIDC;
    const int cvtGrid = (nFeat + 255) / 256;
    const int gemmGrid = (nNodes + 127) / 128;
    const size_t lds1 = 128 * (128 + 8) * sizeof(__bf16);   // 34,816 B
    const size_t lds2 = 128 * (256 + 8) * sizeof(__bf16);   // 67,584 B

    // input activations -> bf16
    f32_to_bf16_kernel<<<cvtGrid, 256, 0, stream>>>(x, xb, nFeat);

    // transpose+convert all weights once
    for (int i = 0; i < 5; ++i) {
        transpose_bf16_kernel<<<(128 * 128 + 255) / 256, 256, 0, stream>>>(Wl[i], wlT[i], 128, 128);
        transpose_bf16_kernel<<<(256 * 128 + 255) / 256, 256, 0, stream>>>(Wu[i], wuT[i], 256, 128);
    }

    const int* srcIdx = edges;
    const int* dstIdx = edges + nEdges;
    const int scatGrid = ((nEdges * 32) + 255) / 256;

    for (int layer = 0; layer < 5; ++layer) {
        // h = bf16(x) @ Wl + bl    -> hF (f32, for scatter) and hB (bf16)
        mpnn_gemm_wmma_kernel<<<gemmGrid, 256, lds1, stream>>>(
            xb, nullptr, wlT[layer], bl[layer], hF, hB, nNodes, 128, 0);

        // m = segment_sum(h[src], dst)
        zero_f32_kernel<<<cvtGrid, 256, 0, stream>>>(mF, nFeat);
        edge_scatter_kernel<<<scatGrid, 256, 0, stream>>>(hF, srcIdx, dstIdx, mF, nEdges);
        f32_to_bf16_kernel<<<cvtGrid, 256, 0, stream>>>(mF, mB, nFeat);

        // x = leaky( [h|m] @ Wu + bu )  -> xb (bf16 next-layer) and hF (f32)
        mpnn_gemm_wmma_kernel<<<gemmGrid, 256, lds2, stream>>>(
            hB, mB, wuT[layer], bu[layer], hF, xb, nNodes, 256, 1);
    }

    // global mean pool + head + log_softmax
    zero_f32_kernel<<<(nGraphs * HIDC + 255) / 256, 256, 0, stream>>>(pooled, nGraphs * HIDC);
    zero_f32_kernel<<<1, 256, 0, stream>>>(cnt, nGraphs);
    pool_kernel<<<((nNodes * 32) + 255) / 256, 256, 0, stream>>>(hF, batch, pooled, cnt, nNodes);
    head_kernel<<<1, 256, 0, stream>>>(pooled, cnt, Wout, bout, (float*)d_out, nGraphs);
}
